// Task2SupervisedModel_15539191677324
// MI455X (gfx1250) — matrix operand order repaired and checked by reading, not compiled, not run on hardware
//
#include <hip/hip_runtime.h>
#include <hip/hip_bf16.h>

// ---------------------------------------------------------------------------
// Hypergraph 2-layer mean-pool conv + link prediction for MI455X (gfx1250).
//  - GEMMs via v_wmma_f32_16x16x32_bf16 (wave32, one 16x16 tile per wave)
//  - segment sums via global f32 atomics (dest-only norms folded out of loop)
//  - everything else f32
// ---------------------------------------------------------------------------

#define N_NODES 50000
#define N_EDGES 25000
#define NNZ_CNT 800000
#define DIM     256
#define N_PAIRS 100000

typedef __attribute__((ext_vector_type(16))) __bf16   v16bf;
typedef __attribute__((ext_vector_type(8)))  float    v8f;
typedef __attribute__((ext_vector_type(8)))  unsigned v8u;

__device__ __forceinline__ __bf16 to_bf16(float f) {
    unsigned u = __builtin_bit_cast(unsigned, f);
    unsigned r = u + 0x7fffu + ((u >> 16) & 1u);       // round-to-nearest-even
    unsigned short hs = (unsigned short)(r >> 16);
    return __builtin_bit_cast(__bf16, hs);
}

__device__ __forceinline__ float prelu_f(float v, float a) {
    return v >= 0.0f ? v : a * v;
}

// ---- f32 -> bf16 bulk convert (4 elems / thread) --------------------------
__global__ void k_f32_to_bf16(const float* __restrict__ in,
                              __bf16* __restrict__ out, long n4) {
    long t = (long)blockIdx.x * blockDim.x + threadIdx.x;
    if (t >= n4) return;
    float4 v = ((const float4*)in)[t];
    __bf16* o = out + 4 * t;
    o[0] = to_bf16(v.x); o[1] = to_bf16(v.y);
    o[2] = to_bf16(v.z); o[3] = to_bf16(v.w);
}

// ---- W[k][n] (f32) -> WT[n][k] (bf16) -------------------------------------
__global__ void k_w_transpose_bf16(const float* __restrict__ W,
                                   __bf16* __restrict__ WT) {
    int t = blockIdx.x * blockDim.x + threadIdx.x;   // 65536 threads
    int n = t >> 8, k = t & 255;
    WT[n * DIM + k] = to_bf16(W[k * DIM + n]);
}

// ---- degree count ----------------------------------------------------------
__global__ void k_degree(const int* __restrict__ nidx, const int* __restrict__ eidx,
                         float* __restrict__ Dn, float* __restrict__ De, int nnz) {
    int t = blockIdx.x * blockDim.x + threadIdx.x;
    if (t >= nnz) return;
    atomicAdd(&Dn[nidx[t]], 1.0f);
    atomicAdd(&De[eidx[t]], 1.0f);
}

__global__ void k_recip(float* __restrict__ d, int n) {
    int t = blockIdx.x * blockDim.x + threadIdx.x;
    if (t >= n) return;
    float v = d[t];
    d[t] = v > 0.0f ? 1.0f / v : 0.0f;
}

// ---- C[M x 256] = A[M x 256](bf16) * W  with W stored transposed (bf16) ---
// one wave per 16x16 output tile, 8 waves per block, K-loop of 8 WMMAs
__global__ void k_gemm_bf16_256(const __bf16* __restrict__ A,
                                const __bf16* __restrict__ WT,
                                float* __restrict__ C, int M) {
    const int lane = threadIdx.x & 31;
    const int wave = threadIdx.x >> 5;
    const int tile = blockIdx.x * 8 + wave;
    const int rowTiles = (M + 15) >> 4;
    if (tile >= rowTiles * 16) return;               // 16 col tiles (N=256)
    const int row_tile = tile >> 4;
    const int col_tile = tile & 15;
    const int half = lane >> 4;
    const int mr   = lane & 15;

    int ra = row_tile * 16 + mr;                     // A row this lane loads
    if (ra > M - 1) ra = M - 1;                      // clamp partial tile
    const uint4* Ap4 = (const uint4*)(A  + (long long)ra * DIM);
    const uint4* Bp4 = (const uint4*)(WT + (long long)(col_tile * 16 + mr) * DIM);

    v8f acc = {};
#pragma unroll
    for (int ks = 0; ks < 8; ++ks) {
        // A lane layout: K = {2v+8*half}_{v<4} , {16+2(v-4)+8*half}_{v>=4}
        uint4 alo = Ap4[4 * ks + half];
        uint4 ahi = Ap4[4 * ks + 2 + half];
        // B lane layout: K = {2v+8*half}_{v<4} , {16+2(v-4)+8*half}_{v>=4}
        uint4 blo = Bp4[4 * ks + half];
        uint4 bhi = Bp4[4 * ks + 2 + half];
        v8u ar; ar[0]=alo.x; ar[1]=alo.y; ar[2]=alo.z; ar[3]=alo.w;
                ar[4]=ahi.x; ar[5]=ahi.y; ar[6]=ahi.z; ar[7]=ahi.w;
        v8u br; br[0]=blo.x; br[1]=blo.y; br[2]=blo.z; br[3]=blo.w;
                br[4]=bhi.x; br[5]=bhi.y; br[6]=bhi.z; br[7]=bhi.w;
        v16bf a = __builtin_bit_cast(v16bf, ar);
        v16bf b = __builtin_bit_cast(v16bf, br);
        acc = __builtin_amdgcn_wmma_f32_16x16x32_bf16(
                  false, a, false, b, (short)0, acc, false, false);
    }
    const int nc = (col_tile << 4) + mr;             // C: lane -> N column
#pragma unroll
    for (int v = 0; v < 8; ++v) {                    // VGPR v -> M = v + 8*half
        int mg = (row_tile << 4) + v + 8 * half;
        if (mg < M) C[(long long)mg * DIM + nc] = acc[v];
    }
}

// ---- dst[didx[e]] += src[sidx[e]]  (64 threads / edge, 4 floats each) -----
__global__ void k_scatter_rows(const float* __restrict__ src,
                               float* __restrict__ dst,
                               const int* __restrict__ sidx,
                               const int* __restrict__ didx, int nnz) {
    long t = (long)blockIdx.x * blockDim.x + threadIdx.x;
    long e = t >> 6;
    if (e >= nnz) return;
    int d = (int)(t & 63) * 4;
    float4 v = *(const float4*)(src + (long long)sidx[e] * DIM + d);
    float* dp = dst + (long long)didx[e] * DIM + d;
    atomicAdd(dp + 0, v.x); atomicAdd(dp + 1, v.y);
    atomicAdd(dp + 2, v.z); atomicAdd(dp + 3, v.w);
}

// ---- edge features: Eb = bf16(prelu(De_inv * acc + bias)) -----------------
__global__ void k_edge_finalize(const float* __restrict__ acc,
                                const float* __restrict__ dinv,
                                const float* __restrict__ bias,
                                const float* __restrict__ a_p,
                                __bf16* __restrict__ out, int M) {
    long t = (long)blockIdx.x * blockDim.x + threadIdx.x;
    long r = t >> 6;
    if (r >= M) return;
    int d = (int)(t & 63) * 4;
    float s = dinv[r], a = *a_p;
    float4 v = *(const float4*)(acc + r * DIM + d);
    float4 b = *(const float4*)(bias + d);
    __bf16* o = out + r * DIM + d;
    o[0] = to_bf16(prelu_f(s * v.x + b.x, a));
    o[1] = to_bf16(prelu_f(s * v.y + b.y, a));
    o[2] = to_bf16(prelu_f(s * v.z + b.z, a));
    o[3] = to_bf16(prelu_f(s * v.w + b.w, a));
}

// ---- node features: h = prelu(Dn_inv * acc + bias)  (f32) -----------------
__global__ void k_node_finalize(const float* __restrict__ acc,
                                const float* __restrict__ dinv,
                                const float* __restrict__ bias,
                                const float* __restrict__ a_p,
                                float* __restrict__ out, int M) {
    long t = (long)blockIdx.x * blockDim.x + threadIdx.x;
    long r = t >> 6;
    if (r >= M) return;
    int d = (int)(t & 63) * 4;
    float s = dinv[r], a = *a_p;
    float4 v = *(const float4*)(acc + r * DIM + d);
    float4 b = *(const float4*)(bias + d);
    float4 o;
    o.x = prelu_f(s * v.x + b.x, a);
    o.y = prelu_f(s * v.y + b.y, a);
    o.z = prelu_f(s * v.z + b.z, a);
    o.w = prelu_f(s * v.w + b.w, a);
    *(float4*)(out + r * DIM + d) = o;
}

// ---- pred = sigmoid(dot(h[i1], h[i2])) ; one wave32 per pair --------------
__global__ void k_pair_dot_sigmoid(const float* __restrict__ h,
                                   const int* __restrict__ ind1,
                                   const int* __restrict__ ind2,
                                   float* __restrict__ out, int npairs) {
    int lane = threadIdx.x & 31;
    int w = (int)(((long)blockIdx.x * blockDim.x + threadIdx.x) >> 5);
    if (w >= npairs) return;
    const float4* p1 = (const float4*)(h + (long long)ind1[w] * DIM);
    const float4* p2 = (const float4*)(h + (long long)ind2[w] * DIM);
    float s = 0.0f;
#pragma unroll
    for (int i = 0; i < 2; ++i) {
        float4 a = p1[lane + 32 * i], b = p2[lane + 32 * i];
        s += a.x * b.x + a.y * b.y + a.z * b.z + a.w * b.w;
    }
#pragma unroll
    for (int off = 16; off > 0; off >>= 1) s += __shfl_down(s, off, 32);
    if (lane == 0) out[w] = 1.0f / (1.0f + __expf(-s));
}

// ---------------------------------------------------------------------------
extern "C" void kernel_launch(void* const* d_in, const int* in_sizes, int n_in,
                              void* d_out, int out_size, void* d_ws, size_t ws_size,
                              hipStream_t stream) {
    const float* x    = (const float*)d_in[0];
    const int*   he   = (const int*)d_in[1];
    const int*   nidx = he;
    const int*   eidx = he + NNZ_CNT;
    const int*   ind1 = (const int*)d_in[4];
    const int*   ind2 = (const int*)d_in[5];
    const float* Wm[4] = { (const float*)d_in[6],  (const float*)d_in[8],
                           (const float*)d_in[10], (const float*)d_in[12] };
    const float* Bv[4] = { (const float*)d_in[7],  (const float*)d_in[9],
                           (const float*)d_in[11], (const float*)d_in[13] };
    const float* a_p = (const float*)d_in[14];

    // ---- workspace layout (aliased; ~167 MB total) ----
    char* base = (char*)d_ws;
    size_t off = 0;
    auto take = [&](size_t bytes) -> char* {
        char* p = base + off;
        off = (off + bytes + 255) & ~(size_t)255;
        return p;
    };
    __bf16* WbT[4];
    for (int i = 0; i < 4; ++i) WbT[i] = (__bf16*)take((size_t)DIM * DIM * 2);
    float*  Dn   = (float*)take((size_t)N_NODES * 4);
    float*  De   = (float*)take((size_t)N_EDGES * 4);
    __bf16* Hb   = (__bf16*)take((size_t)N_NODES * DIM * 2);   // bf16 GEMM input
    float*  BigF = (float*)take((size_t)N_NODES * DIM * 4);    // H-gemm out / Nacc
    float*  MedF = (float*)take((size_t)N_EDGES * DIM * 4);    // Eacc / G-gemm out
    __bf16* Eb   = (__bf16*)take((size_t)N_EDGES * DIM * 2);   // bf16 edge features
    float*  Hout = (float*)take((size_t)N_NODES * DIM * 4);    // layer output

    // ---- weights -> bf16 transposed ----
    for (int i = 0; i < 4; ++i)
        k_w_transpose_bf16<<<256, 256, 0, stream>>>(Wm[i], WbT[i]);

    // ---- degrees + reciprocals ----
    hipMemsetAsync(Dn, 0, (size_t)N_NODES * 4, stream);
    hipMemsetAsync(De, 0, (size_t)N_EDGES * 4, stream);
    k_degree<<<(NNZ_CNT + 255) / 256, 256, 0, stream>>>(nidx, eidx, Dn, De, NNZ_CNT);
    k_recip<<<(N_NODES + 255) / 256, 256, 0, stream>>>(Dn, N_NODES);
    k_recip<<<(N_EDGES + 255) / 256, 256, 0, stream>>>(De, N_EDGES);

    const int gemmN = ((((N_NODES + 15) / 16) * 16) + 7) / 8;   // 6250
    const int gemmE = ((((N_EDGES + 15) / 16) * 16) + 7) / 8;   // 3126
    const long n4N = (long)N_NODES * (DIM / 4);                 // 3.2M

    for (int lay = 0; lay < 2; ++lay) {
        const float* hin = (lay == 0) ? x : Hout;
        // 1) node features -> bf16
        k_f32_to_bf16<<<(int)((n4N + 255) / 256), 256, 0, stream>>>(hin, Hb, n4N);
        // 2) H = h @ W_n2e  (WMMA)
        k_gemm_bf16_256<<<gemmN, 256, 0, stream>>>(Hb, WbT[2 * lay], BigF, N_NODES);
        // 3) scatter node rows into edge accumulator
        hipMemsetAsync(MedF, 0, (size_t)N_EDGES * DIM * 4, stream);
        k_scatter_rows<<<(NNZ_CNT * 64) / 256, 256, 0, stream>>>(BigF, MedF, nidx, eidx, NNZ_CNT);
        // 4) e = prelu(De_inv*acc + b) -> bf16
        k_edge_finalize<<<(N_EDGES * 64) / 256, 256, 0, stream>>>(MedF, De, Bv[2 * lay], a_p, Eb, N_EDGES);
        // 5) G = e @ W_e2n  (WMMA)  (writes MedF; Eacc consumed)
        k_gemm_bf16_256<<<gemmE, 256, 0, stream>>>(Eb, WbT[2 * lay + 1], MedF, N_EDGES);
        // 6) scatter edge rows into node accumulator (aliases BigF)
        hipMemsetAsync(BigF, 0, (size_t)N_NODES * DIM * 4, stream);
        k_scatter_rows<<<(NNZ_CNT * 64) / 256, 256, 0, stream>>>(MedF, BigF, eidx, nidx, NNZ_CNT);
        // 7) h = prelu(Dn_inv*acc + b)
        k_node_finalize<<<(N_NODES * 64) / 256, 256, 0, stream>>>(BigF, Dn, Bv[2 * lay + 1], a_p, Hout, N_NODES);
    }

    // ---- link prediction ----
    k_pair_dot_sigmoid<<<(N_PAIRS * 32) / 256, 256, 0, stream>>>(
        Hout, ind1, ind2, (float*)d_out, N_PAIRS);
}
